// inference_model_19387482374819
// MI455X (gfx1250) — matrix operand order repaired and checked by reading, not compiled, not run on hardware
//
#include <hip/hip_runtime.h>
#include <hip/hip_bf16.h>

#define E_  5
#define L_  128
#define H_  768
#define KMU 21
#define NL_ 3

typedef __attribute__((ext_vector_type(16))) _Float16 v16h;
typedef __attribute__((ext_vector_type(8)))  _Float16 v8h;
typedef __attribute__((ext_vector_type(8)))  float    v8f;

// ---------------------------------------------------------------------------
// Fragment loaders per CDNA5 ISA 16-bit WMMA layouts (CDNA5 ISA):
// A 16x32 (MxK): lane half 0 holds K in [h0+0,8)∪[h0+16,24); half 1 holds
//                K in [h0+8,16)∪[h0+24,32). Two 16B loads per lane.
// B 32x16 (KxN): same K split as A: half 0 holds K in [h0+0,8)∪[h0+16,24);
//                half 1 holds K in [h0+8,16)∪[h0+24,32). Two 16B loads per lane.
// Both operands come from row-major (seq, H) arrays.
// ---------------------------------------------------------------------------
__device__ __forceinline__ v16h load_frag_a(const _Float16* rowp, int h0, int half) {
    const _Float16* p = rowp + h0 + half * 8;
    v8h a = *(const v8h*)p;
    v8h b = *(const v8h*)(p + 16);
    v16h r;
#pragma unroll
    for (int t = 0; t < 8; ++t) { r[t] = a[t]; r[t + 8] = b[t]; }
    return r;
}

__device__ __forceinline__ v16h load_frag_b(const _Float16* rowp, int h0, int half) {
    const _Float16* p = rowp + h0 + half * 8;
    v8h a = *(const v8h*)p;
    v8h b = *(const v8h*)(p + 16);
    v16h r;
#pragma unroll
    for (int t = 0; t < 8; ++t) { r[t] = a[t]; r[t + 8] = b[t]; }
    return r;
}

__device__ __forceinline__ v8f wmma_f16(v16h a, v16h b, v8f c) {
    return __builtin_amdgcn_wmma_f32_16x16x32_f16(false, a, false, b, (short)0, c,
                                                  false, false);
}

// ---------------------------------------------------------------------------
// Kernel 1: per-(n,l) L2 normalize over H, split to f16 hi + f16 residual lo.
// ---------------------------------------------------------------------------
__global__ void __launch_bounds__(256) k_norm_split(
    const float* __restrict__ x, _Float16* __restrict__ hi, _Float16* __restrict__ lo)
{
    size_t row = blockIdx.x;
    const float* p = x + row * H_;
    int tid = threadIdx.x;
    float s = 0.f;
    for (int h = tid; h < H_; h += 256) { float v = p[h]; s += v * v; }
    __shared__ float red[256];
    red[tid] = s;
    __syncthreads();
    for (int st = 128; st > 0; st >>= 1) {
        if (tid < st) red[tid] += red[tid + st];
        __syncthreads();
    }
    float inv = 1.0f / fmaxf(sqrtf(red[0]), 1e-12f);
    for (int h = tid; h < H_; h += 256) {
        float v = p[h] * inv;
        _Float16 a = (_Float16)v;
        float r = v - (float)a;
        hi[row * H_ + h] = a;
        lo[row * H_ + h] = (_Float16)r;
    }
}

// ---------------------------------------------------------------------------
// Kernel 2: WMMA gram (hi/lo split f16 -> f32-accurate) fused with 21-channel
// Gaussian key pooling and log-score. One block per (b, e, q-tile, i);
// 4 waves, each wave owns two 16x16 output tiles over keys.
// ---------------------------------------------------------------------------
__global__ void __launch_bounds__(128) k_gram_pool(
    const _Float16* __restrict__ hn_hi, const _Float16* __restrict__ hn_lo,
    const int* __restrict__ msk, const float* __restrict__ w_att,
    const float* __restrict__ b_att, float* __restrict__ score, int B)
{
    int x  = blockIdx.x;
    int qt = x & 7;  x >>= 3;
    int i  = x % E_;
    int e  = x / E_;
    int b  = blockIdx.y;

    int lane = threadIdx.x & 31;
    int wid  = threadIdx.x >> 5;
    int m    = lane & 15;
    int half = lane >> 4;

    __shared__ float S_w[4][16][KMU + 1];   // per-wave partial sums (deterministic)

    size_t arow  = ((size_t)(b * E_ + e) * L_ + qt * 16) * H_;
    int    n0a   = wid * 32;
    int    n0b   = wid * 32 + 16;
    size_t brow0 = ((size_t)(b * E_ + i) * L_ + n0a) * H_;
    size_t brow1 = ((size_t)(b * E_ + i) * L_ + n0b) * H_;

    const _Float16* AH  = hn_hi + arow  + (size_t)m * H_;
    const _Float16* AL  = hn_lo + arow  + (size_t)m * H_;
    const _Float16* B0H = hn_hi + brow0 + (size_t)m * H_;
    const _Float16* B0L = hn_lo + brow0 + (size_t)m * H_;
    const _Float16* B1H = hn_hi + brow1 + (size_t)m * H_;
    const _Float16* B1L = hn_lo + brow1 + (size_t)m * H_;

    v8f acc0 = {}, acc1 = {};
    for (int kc = 0; kc < H_ / 32; ++kc) {
        int h0 = kc * 32;
        v16h aH  = load_frag_a(AH,  h0, half);
        v16h aL  = load_frag_a(AL,  h0, half);
        v16h b0H = load_frag_b(B0H, h0, half);
        v16h b0L = load_frag_b(B0L, h0, half);
        v16h b1H = load_frag_b(B1H, h0, half);
        v16h b1L = load_frag_b(B1L, h0, half);
        // hi*lo + lo*hi + hi*hi  (lo*lo term negligible, ~2^-22)
        acc0 = wmma_f16(aL, b0H, acc0);
        acc0 = wmma_f16(aH, b0L, acc0);
        acc0 = wmma_f16(aH, b0H, acc0);
        acc1 = wmma_f16(aL, b1H, acc1);
        acc1 = wmma_f16(aH, b1L, acc1);
        acc1 = wmma_f16(aH, b1H, acc1);
    }

    // key masks: mask_text[b*E+i][k] = (k>0) && msk!=0
    int mrow = (b * E_ + i) * L_;
    int k0 = n0a + m, k1 = n0b + m;
    float km0 = (k0 > 0 && msk[mrow + k0] != 0) ? 1.f : 0.f;
    float km1 = (msk[mrow + k1] != 0) ? 1.f : 0.f;

    for (int j = 0; j < KMU; ++j) {
        float mu    = (j == 0) ? 1.0f  : (0.95f - 0.1f * (float)(j - 1));
        float inv2s = (j == 0) ? 5.0e5f : 50.0f;   // 1/(2*sig^2)
#pragma unroll
        for (int v = 0; v < 8; ++v) {
            float d0 = acc0[v] - mu;
            float d1 = acc1[v] - mu;
            float p  = __expf(-d0 * d0 * inv2s) * km0
                     + __expf(-d1 * d1 * inv2s) * km1;
            // sum over 16 keys held by each 16-lane group
            p += __shfl_xor(p, 1, 16);
            p += __shfl_xor(p, 2, 16);
            p += __shfl_xor(p, 4, 16);
            p += __shfl_xor(p, 8, 16);
            if (m == 0) S_w[wid][v + 8 * half][j] = p;
        }
    }
    __syncthreads();

    if (threadIdx.x < 16) {
        int q = threadIdx.x;
        float sc = b_att[0];
        for (int j = 0; j < KMU; ++j) {
            float s = S_w[0][q][j] + S_w[1][q][j] + S_w[2][q][j] + S_w[3][q][j];
            sc += logf(fmaxf(s, 1e-10f)) * w_att[j];
        }
        score[((size_t)(b * E_ + e) * L_ + qt * 16 + q) * E_ + i] = sc;
    }
}

// ---------------------------------------------------------------------------
// Kernel 3: masked softmax over L (per b,e,i) and den[b,e,:] = att @ hiddens.
// ---------------------------------------------------------------------------
__global__ void __launch_bounds__(256) k_att_den(
    const float* __restrict__ score, const int* __restrict__ msk,
    const int* __restrict__ seg, const float* __restrict__ hiddens,
    float* __restrict__ den, int B)
{
    int e = blockIdx.x / E_, i = blockIdx.x % E_;
    int b = blockIdx.y;
    int tid = threadIdx.x;
    int row = b * E_ + e;

    __shared__ float att[L_];
    __shared__ float red[256];

    float my = -1e30f;
    if (tid < L_) {
        int l = tid;
        bool mk = (l > 0) && (msk[row * L_ + l] != 0) && (seg[row * L_ + l] != 0);
        float sc = score[((size_t)row * L_ + l) * E_ + i];
        my = mk ? sc : -10000.0f;
    }
    red[tid] = my;
    __syncthreads();
    for (int st = 128; st > 0; st >>= 1) {
        if (tid < st) red[tid] = fmaxf(red[tid], red[tid + st]);
        __syncthreads();
    }
    float mx = red[0];
    __syncthreads();
    float ex = (tid < L_) ? __expf(my - mx) : 0.f;
    red[tid] = ex;
    __syncthreads();
    for (int st = 128; st > 0; st >>= 1) {
        if (tid < st) red[tid] += red[tid + st];
        __syncthreads();
    }
    float denom = red[0];
    if (tid < L_) att[tid] = ex / denom;
    __syncthreads();

    const float* hb = hiddens + (size_t)row * L_ * H_;
    float* dn = den + ((size_t)(i * B + b) * E_ + e) * H_;
    for (int h = tid; h < H_; h += 256) {
        float a = 0.f;
        for (int l = 0; l < L_; ++l) a += att[l] * hb[(size_t)l * H_ + h];
        dn[h] = a;
    }
}

// ---------------------------------------------------------------------------
// Kernel 4: gating MLP (1536->128->1), softmax over E, combine den -> att_de.
// One block per (i, b).
// ---------------------------------------------------------------------------
__global__ void __launch_bounds__(128) k_gate(
    const float* __restrict__ pooled, const float* __restrict__ den,
    const float* __restrict__ w1, const float* __restrict__ b1,
    const float* __restrict__ w2, const float* __restrict__ b2,
    float* __restrict__ att_de, int B)
{
    int i = blockIdx.x, b = blockIdx.y, tid = threadIdx.x;
    __shared__ float s_e[E_];
    __shared__ float red[128];

    const float* own = pooled + (size_t)(b * E_ + i) * H_;
    for (int e = 0; e < E_; ++e) {
        const float* dn = den + ((size_t)(i * B + b) * E_ + e) * H_;
        float acc = b1[tid];
        for (int c = 0; c < H_; ++c) acc += own[c] * w1[c * 128 + tid];
        for (int c = 0; c < H_; ++c) acc += dn[c]  * w1[(H_ + c) * 128 + tid];
        float h = fmaxf(acc, 0.f);
        red[tid] = h * w2[tid];
        __syncthreads();
        for (int st = 64; st > 0; st >>= 1) {
            if (tid < st) red[tid] += red[tid + st];
            __syncthreads();
        }
        if (tid == 0) s_e[e] = red[0] + b2[0];
        __syncthreads();
    }

    float mx = s_e[0];
    for (int e = 1; e < E_; ++e) mx = fmaxf(mx, s_e[e]);
    float wgt[E_];
    float ws = 0.f;
    for (int e = 0; e < E_; ++e) { wgt[e] = __expf(s_e[e] - mx); ws += wgt[e]; }
    for (int e = 0; e < E_; ++e) wgt[e] /= ws;

    for (int h = tid; h < H_; h += 128) {
        float o = 0.f;
        for (int e = 0; e < E_; ++e)
            o += wgt[e] * den[((size_t)(i * B + b) * E_ + e) * H_ + h];
        att_de[(size_t)(b * E_ + i) * H_ + h] = o;
    }
}

// ---------------------------------------------------------------------------
// Kernel 5: de = tanh(pb @ w_de + b_de) * att_de; feat = [pb,de] @ w_inf + b_inf.
// One block per (e, b).
// ---------------------------------------------------------------------------
__global__ void __launch_bounds__(256) k_de_feat(
    const float* __restrict__ pooled, const float* __restrict__ att_de,
    const float* __restrict__ w_de, const float* __restrict__ b_de,
    const float* __restrict__ w_inf, const float* __restrict__ b_inf,
    float* __restrict__ feat, int B)
{
    int e = blockIdx.x, b = blockIdx.y, tid = threadIdx.x;
    int row = b * E_ + e;
    const float* pb = pooled + (size_t)row * H_;

    float p0 = 0.f, p1 = 0.f, p2 = 0.f;
    for (int c = tid; c < H_; c += 256) {
        float v = pb[c];
        p0 += v * w_inf[c * NL_ + 0];
        p1 += v * w_inf[c * NL_ + 1];
        p2 += v * w_inf[c * NL_ + 2];
    }
    for (int ho = tid; ho < H_; ho += 256) {
        float t = b_de[ho];
        for (int c = 0; c < H_; ++c) t += pb[c] * w_de[(size_t)c * H_ + ho];
        float d = tanhf(t) * att_de[(size_t)row * H_ + ho];
        int cc = H_ + ho;
        p0 += d * w_inf[cc * NL_ + 0];
        p1 += d * w_inf[cc * NL_ + 1];
        p2 += d * w_inf[cc * NL_ + 2];
    }

    __shared__ float red[256];
    float parts[3] = { p0, p1, p2 };
    for (int n = 0; n < 3; ++n) {
        red[tid] = parts[n];
        __syncthreads();
        for (int st = 128; st > 0; st >>= 1) {
            if (tid < st) red[tid] += red[tid + st];
            __syncthreads();
        }
        if (tid == 0) feat[(size_t)row * NL_ + n] = red[0] + b_inf[n];
        __syncthreads();
    }
}

// ---------------------------------------------------------------------------
// Kernel 6: prob = log(mean_e softmax_NL(feat)).
// ---------------------------------------------------------------------------
__global__ void __launch_bounds__(256) k_final(
    const float* __restrict__ feat, float* __restrict__ out, int B)
{
    __shared__ float p[1024 * NL_];
    int tid = threadIdx.x;
    int NE = B * E_;
    for (int t = tid; t < NE; t += 256) {
        float f0 = feat[t * NL_ + 0], f1 = feat[t * NL_ + 1], f2 = feat[t * NL_ + 2];
        float mx = fmaxf(f0, fmaxf(f1, f2));
        float e0 = __expf(f0 - mx), e1 = __expf(f1 - mx), e2 = __expf(f2 - mx);
        float s = e0 + e1 + e2;
        p[t * NL_ + 0] = e0 / s;
        p[t * NL_ + 1] = e1 / s;
        p[t * NL_ + 2] = e2 / s;
    }
    __syncthreads();
    for (int t = tid; t < B * NL_; t += 256) {
        int b = t / NL_, n = t % NL_;
        float a = 0.f;
        for (int e = 0; e < E_; ++e) a += p[(b * E_ + e) * NL_ + n];
        out[t] = logf(a / (float)E_);
    }
}

// ---------------------------------------------------------------------------
extern "C" void kernel_launch(void* const* d_in, const int* in_sizes, int n_in,
                              void* d_out, int out_size, void* d_ws, size_t ws_size,
                              hipStream_t stream)
{
    const float* hiddens = (const float*)d_in[0];
    const float* pooled  = (const float*)d_in[1];
    const int*   msk     = (const int*)d_in[2];
    const int*   seg     = (const int*)d_in[3];
    const float* w_att   = (const float*)d_in[4];
    const float* b_att   = (const float*)d_in[5];
    const float* w_gat1  = (const float*)d_in[6];
    const float* b_gat1  = (const float*)d_in[7];
    const float* w_gat2  = (const float*)d_in[8];
    const float* b_gat2  = (const float*)d_in[9];
    const float* w_de    = (const float*)d_in[10];
    const float* b_de    = (const float*)d_in[11];
    const float* w_inf   = (const float*)d_in[12];
    const float* b_inf   = (const float*)d_in[13];

    int N = in_sizes[0] / (L_ * H_);   // 160
    int B = N / E_;                    // 32

    // workspace carve-out
    size_t nlh = (size_t)N * L_ * H_;
    _Float16* hn_hi = (_Float16*)d_ws;
    _Float16* hn_lo = hn_hi + nlh;
    float* score  = (float*)(hn_lo + nlh);                 // N*L*E
    float* den    = score + (size_t)N * L_ * E_;           // E*B*E*H
    float* att_de = den + (size_t)E_ * B * E_ * H_;        // N*H
    float* feat   = att_de + (size_t)N * H_;               // N*NL
    float* out    = (float*)d_out;

    k_norm_split<<<N * L_, 256, 0, stream>>>(hiddens, hn_hi, hn_lo);
    k_gram_pool<<<dim3(8 * E_ * E_, B), 128, 0, stream>>>(hn_hi, hn_lo, msk,
                                                          w_att, b_att, score, B);
    k_att_den<<<dim3(E_ * E_, B), 256, 0, stream>>>(score, msk, seg, hiddens, den, B);
    k_gate<<<dim3(E_, B), 128, 0, stream>>>(pooled, den, w_gat1, b_gat1,
                                            w_gat2, b_gat2, att_de, B);
    k_de_feat<<<dim3(E_, B), 256, 0, stream>>>(pooled, att_de, w_de, b_de,
                                               w_inf, b_inf, feat, B);
    k_final<<<1, 256, 0, stream>>>(feat, out, B);
}
